// SubstructureAwareDeformableAttention_81509889343826
// MI455X (gfx1250) — matrix operand order repaired and checked by reading, not compiled, not run on hardware
//
#include <hip/hip_runtime.h>
#include <hip/hip_bf16.h>
#include <stdint.h>

// ---------------- problem constants (match reference) ----------------
constexpr int Bc  = 8;
constexpr int Pc  = 2048;
constexpr int Ec  = 256;
constexpr int Hc  = 8;
constexpr int Sc  = 8;
constexpr int HDc = 32;          // Ec / Hc  == wave32 width
constexpr int GHc = 64;
constexpr int GWc = 64;
constexpr int Mtok = Bc * Pc;    // 16384 token rows
constexpr int Msmp = Mtok * Sc;  // 131072 sample rows

typedef _Float16 half_t;
typedef __attribute__((ext_vector_type(16))) _Float16 v16h;
typedef __attribute__((ext_vector_type(8)))  float    v8f;

union Frag16 {
  v16h     h;
  uint32_t u[8];
  uint4    q[2];
};

__device__ __forceinline__ float gelu_exact(float x) {
  return 0.5f * x * (1.0f + erff(x * 0.7071067811865476f));
}

// ---------------- weight packing into WMMA B-fragment layout ----------------
// Layout: [nTile][kStep][lane 0..31][half 0..15], lane%16 = N column,
// half i covers K = kStep*32 + (i&7) + (i>=8 ? 16 : 0) + (lane>=16 ? 8 : 0).
__global__ void pack_weight_kernel(const float* __restrict__ w,
                                   half_t* __restrict__ wp, int K, int N) {
  int total = N * K; // (N/16)*(K/32)*512
  int idx = blockIdx.x * 256 + threadIdx.x;
  if (idx >= total) return;
  int i    = idx & 15;
  int lane = (idx >> 4) & 31;
  int ks   = (idx >> 9) % (K >> 5);
  int nt   = (idx >> 9) / (K >> 5);
  int n = (nt << 4) + (lane & 15);
  int k = (ks << 5) + (i & 7) + ((i >> 3) << 4) + ((lane >= 16) ? 8 : 0);
  wp[idx] = (half_t)w[(size_t)k * N + n];
}

// ---------------- layernorm: f32 in -> f16 out, D = blockDim.x (<=512) ------
__global__ void layernorm_kernel(const float* __restrict__ in,
                                 const float* __restrict__ g,
                                 const float* __restrict__ b,
                                 half_t* __restrict__ out, int D) {
  __shared__ float red[512];
  int row = blockIdx.x, t = threadIdx.x;
  float v = in[(size_t)row * D + t];
  red[t] = v; __syncthreads();
  for (int s = blockDim.x >> 1; s > 0; s >>= 1) {
    if (t < s) red[t] += red[t + s];
    __syncthreads();
  }
  float mean = red[0] / (float)D;
  __syncthreads();
  float d = v - mean;
  red[t] = d * d; __syncthreads();
  for (int s = blockDim.x >> 1; s > 0; s >>= 1) {
    if (t < s) red[t] += red[t + s];
    __syncthreads();
  }
  float inv = rsqrtf(red[0] / (float)D + 1e-5f);
  out[(size_t)row * D + t] = (half_t)(d * inv * g[t] + b[t]);
}

// ---------------- generic WMMA GEMM: C[M,N] = A[M,K] * Wp + bias ------------
// Wave tile = 16(M) x 64(N). Software-pipelined K loop (ping/pong fragment
// registers) so the 4 WMMAs of one stage overlap the 10 b128 loads of the
// next stage.
// mode 0: bias only
// mode 1: bias + exact gelu
// mode 2: val = res + (pad ? 0 : val)           (o-projection residual)
// mode 3: val = pad ? 0 : (res + val)           (final output)
__global__ void __launch_bounds__(256)
gemm_wmma_kernel(const half_t* __restrict__ A, const half_t* __restrict__ Wp,
                 const float* __restrict__ bias,
                 float* out32, half_t* out16,
                 const float* __restrict__ res,
                 const unsigned char* __restrict__ pad,
                 int M, int N, int K, int mode) {
  const int wave = threadIdx.x >> 5;
  const int lane = threadIdx.x & 31;
  const int nGroups = N >> 6;               // groups of 4 n-tiles (64 cols)
  int wtile = blockIdx.x * 8 + wave;
  if (wtile >= (M >> 4) * nGroups) return;
  const int nG = wtile % nGroups;
  const int mT = wtile / nGroups;
  const int m0 = mT << 4;
  const int n0 = nG << 6;
  const int kSteps = K >> 5;                // always even here (8 or 16)

  // A fragment addressing (16-bit A 16x32 layout, ISA 7.12.2):
  // lane%16 = M row; lanes>=16 use K+8 within each half-group.
  const int mRow = m0 + (lane & 15);
  const int kHalfSel = (lane >> 4) << 3;    // 0 or 8
  const half_t* aRow = A + (size_t)mRow * K;
  const uint32_t* wb0 =
      (const uint32_t*)Wp + ((size_t)(nG * 4 + 0) * kSteps) * 256 + (size_t)lane * 8;
  const uint32_t* wb1 = wb0 + (size_t)kSteps * 256;
  const uint32_t* wb2 = wb1 + (size_t)kSteps * 256;
  const uint32_t* wb3 = wb2 + (size_t)kSteps * 256;

  auto load_a = [&](Frag16& a, int ks) {
    const uint32_t* ar = (const uint32_t*)(aRow + (ks << 5));
#pragma unroll
    for (int p = 0; p < 8; ++p) {
      int k = ((p < 4) ? 0 : 16) + ((p & 3) << 1) + kHalfSel;
      a.u[p] = ar[k >> 1];
    }
  };
  auto load_b = [&](Frag16* b, int ks) {
    const uint4* q0 = (const uint4*)(wb0 + (size_t)ks * 256);
    const uint4* q1 = (const uint4*)(wb1 + (size_t)ks * 256);
    const uint4* q2 = (const uint4*)(wb2 + (size_t)ks * 256);
    const uint4* q3 = (const uint4*)(wb3 + (size_t)ks * 256);
    b[0].q[0] = q0[0]; b[0].q[1] = q0[1];
    b[1].q[0] = q1[0]; b[1].q[1] = q1[1];
    b[2].q[0] = q2[0]; b[2].q[1] = q2[1];
    b[3].q[0] = q3[0]; b[3].q[1] = q3[1];
  };

  v8f acc[4] = {};
  Frag16 aP, bP[4], aQ, bQ[4];
  load_a(aP, 0);
  load_b(bP, 0);
  for (int ks = 0; ks < kSteps; ks += 2) {
    // stage Q loads fly while stage P WMMAs execute
    if (ks + 1 < kSteps) {
      __builtin_prefetch(aRow + ((ks + 1) << 5), 0, 3);
      load_a(aQ, ks + 1);
      load_b(bQ, ks + 1);
    }
#pragma unroll
    for (int j = 0; j < 4; ++j)
      acc[j] = __builtin_amdgcn_wmma_f32_16x16x32_f16(
          false, aP.h, false, bP[j].h, (short)0, acc[j], false, false);
    if (ks + 2 < kSteps) {
      __builtin_prefetch(aRow + ((ks + 2) << 5), 0, 3);
      load_a(aP, ks + 2);
      load_b(bP, ks + 2);
    }
#pragma unroll
    for (int j = 0; j < 4; ++j)
      acc[j] = __builtin_amdgcn_wmma_f32_16x16x32_f16(
          false, aQ.h, false, bQ[j].h, (short)0, acc[j], false, false);
  }

  // C/D layout: lane%16 = N col; VGPR r -> M = r + (lane>=16 ? 8 : 0).
  const int mBase = m0 + ((lane >> 4) << 3);
#pragma unroll
  for (int j = 0; j < 4; ++j) {
    const int nCol = n0 + (j << 4) + (lane & 15);
    const float bn = bias ? bias[nCol] : 0.0f;
#pragma unroll
    for (int r = 0; r < 8; ++r) {
      int m = mBase + r;
      float val = acc[j][r] + bn;
      if (mode == 1) {
        val = gelu_exact(val);
      } else if (mode == 2) {
        float o = (pad && pad[m]) ? 0.0f : val;
        val = res[(size_t)m * N + nCol] + o;
      } else if (mode == 3) {
        val = (pad && pad[m]) ? 0.0f : (res[(size_t)m * N + nCol] + val);
      }
      size_t oi = (size_t)m * N + nCol;
      if (out32) out32[oi] = val;
      if (out16) out16[oi] = (half_t)val;
    }
  }
}

// ---------------- offset head: per (token, s) matvec + tanh + clip ----------
__global__ void offsets_kernel(const half_t* __restrict__ h16,
                               const float* __restrict__ w2,
                               const float* __restrict__ b2,
                               const float* __restrict__ base,
                               float* __restrict__ coords,
                               float* __restrict__ dist) {
  int idx = blockIdx.x * blockDim.x + threadIdx.x; // token*S + s
  if (idx >= Mtok * Sc) return;
  int s = idx & (Sc - 1);
  int tok = idx >> 3;
  const half_t* hr = h16 + (size_t)tok * Ec;
  float ox = b2[2 * s], oy = b2[2 * s + 1];
  for (int e = 0; e < Ec; ++e) {
    float hv = (float)hr[e];
    ox = fmaf(hv, w2[e * (Sc * 2) + 2 * s], ox);
    oy = fmaf(hv, w2[e * (Sc * 2) + 2 * s + 1], oy);
  }
  const float offset_scale = 2.0f;
  const float stepx = offset_scale * 2.0f / (float)(GWc - 1);
  const float stepy = offset_scale * 2.0f / (float)(GHc - 1);
  float bx = base[(size_t)tok * 2], by = base[(size_t)tok * 2 + 1];
  float cx = fminf(1.0f, fmaxf(-1.0f, bx + tanhf(ox) * stepx));
  float cy = fminf(1.0f, fmaxf(-1.0f, by + tanhf(oy) * stepy));
  coords[(size_t)idx * 2 + 0] = cx;
  coords[(size_t)idx * 2 + 1] = cy;
  float dx = cx - bx, dy = cy - by;
  dist[idx] = dx * dx + dy * dy;
}

// ---------------- bilinear grid sample, align_corners=true, zero pad --------
// block = token (b,p); thread = channel c; corner data staged in LDS.
// The whole grid (32 MB) is L2-resident, so the gathers are L2 hits.
__global__ void __launch_bounds__(256)
grid_sample_kernel(const float* __restrict__ grid,
                   const float* __restrict__ coords,
                   half_t* __restrict__ samp) {
  __shared__ int   sidx[Sc][4];
  __shared__ float swgt[Sc][4];
  int tok = blockIdx.x;
  int c = threadIdx.x;
  int b = tok / Pc;
  if (c < Sc) {
    int s = c;
    float cx = coords[((size_t)tok * Sc + s) * 2 + 0];
    float cy = coords[((size_t)tok * Sc + s) * 2 + 1];
    float ix = (cx + 1.0f) * 0.5f * (float)(GWc - 1);
    float iy = (cy + 1.0f) * 0.5f * (float)(GHc - 1);
    float x0f = floorf(ix), y0f = floorf(iy);
    float wx = ix - x0f, wy = iy - y0f;
    int x0 = (int)x0f, y0 = (int)y0f;
    int x1 = x0 + 1, y1 = y0 + 1;
    bool vx0 = (x0 >= 0 && x0 < GWc), vy0 = (y0 >= 0 && y0 < GHc);
    bool vx1 = (x1 >= 0 && x1 < GWc), vy1 = (y1 >= 0 && y1 < GHc);
    int x0c = min(max(x0, 0), GWc - 1), x1c = min(max(x1, 0), GWc - 1);
    int y0c = min(max(y0, 0), GHc - 1), y1c = min(max(y1, 0), GHc - 1);
    sidx[s][0] = y0c * GWc + x0c;
    sidx[s][1] = y0c * GWc + x1c;
    sidx[s][2] = y1c * GWc + x0c;
    sidx[s][3] = y1c * GWc + x1c;
    swgt[s][0] = (vx0 && vy0) ? (1.0f - wx) * (1.0f - wy) : 0.0f;
    swgt[s][1] = (vx1 && vy0) ? wx * (1.0f - wy) : 0.0f;
    swgt[s][2] = (vx0 && vy1) ? (1.0f - wx) * wy : 0.0f;
    swgt[s][3] = (vx1 && vy1) ? wx * wy : 0.0f;
  }
  __syncthreads();
  const float* gch = grid + ((size_t)b * Ec + c) * (GHc * GWc);
  size_t ob = (size_t)tok * Sc * Ec + c;
#pragma unroll
  for (int s = 0; s < Sc; ++s) {
    float v = swgt[s][0] * gch[sidx[s][0]] + swgt[s][1] * gch[sidx[s][1]] +
              swgt[s][2] * gch[sidx[s][2]] + swgt[s][3] * gch[sidx[s][3]];
    samp[ob + (size_t)s * Ec] = (half_t)v;
  }
}

// ---------------- attention: HD==32 -> one wave per head --------------------
__global__ void __launch_bounds__(256)
attn_kernel(const half_t* __restrict__ q16, const half_t* __restrict__ k16,
            const half_t* __restrict__ v16, const float* __restrict__ dist,
            const float* __restrict__ dist_pen,
            const unsigned char* __restrict__ pad,
            half_t* __restrict__ out16) {
  int tok = blockIdx.x;
  int h = threadIdx.x >> 5;
  int lane = threadIdx.x & 31;
  int d = h * HDc + lane;
  const float rs = 0.17677669529663687f; // 1/sqrt(32)
  float pen = fabsf(dist_pen[0]);
  bool isPad = pad[tok] != 0;

  float qv = (float)q16[(size_t)tok * Ec + d];
  float logits[Sc];
#pragma unroll
  for (int s = 0; s < Sc; ++s) {
    float kv = (float)k16[((size_t)tok * Sc + s) * Ec + d];
    float p = qv * kv;
#pragma unroll
    for (int m = 16; m >= 1; m >>= 1) p += __shfl_xor(p, m, 32);
    float l = p * rs - pen * dist[(size_t)tok * Sc + s];
    logits[s] = isPad ? 0.0f : l;
  }
  float mx = logits[0];
#pragma unroll
  for (int s = 1; s < Sc; ++s) mx = fmaxf(mx, logits[s]);
  float ex[Sc], sum = 0.0f;
#pragma unroll
  for (int s = 0; s < Sc; ++s) { ex[s] = __expf(logits[s] - mx); sum += ex[s]; }
  float inv = isPad ? 0.0f : (1.0f / sum);
  float o = 0.0f;
#pragma unroll
  for (int s = 0; s < Sc; ++s) {
    float vv = (float)v16[((size_t)tok * Sc + s) * Ec + d];
    o = fmaf(ex[s] * inv, vv, o);
  }
  out16[(size_t)tok * Ec + d] = (half_t)o;
}

// ---------------- host-side launch ------------------------------------------
static inline size_t align_up(size_t x) { return (x + 255) & ~(size_t)255; }

extern "C" void kernel_launch(void* const* d_in, const int* in_sizes, int n_in,
                              void* d_out, int out_size, void* d_ws, size_t ws_size,
                              hipStream_t stream) {
  (void)in_sizes; (void)n_in; (void)out_size; (void)ws_size;
  const float* x        = (const float*)d_in[0];
  const float* grid     = (const float*)d_in[1];
  const float* base     = (const float*)d_in[2];
  const float* ln1_g    = (const float*)d_in[3];
  const float* ln1_b    = (const float*)d_in[4];
  const float* off_w1   = (const float*)d_in[5];
  const float* off_b1   = (const float*)d_in[6];
  const float* off_w2   = (const float*)d_in[7];
  const float* off_b2   = (const float*)d_in[8];
  const float* q_w      = (const float*)d_in[9];
  const float* q_b      = (const float*)d_in[10];
  const float* k_w      = (const float*)d_in[11];
  const float* k_b      = (const float*)d_in[12];
  const float* v_w      = (const float*)d_in[13];
  const float* v_b      = (const float*)d_in[14];
  const float* o_w      = (const float*)d_in[15];
  const float* o_b      = (const float*)d_in[16];
  const float* dist_pen = (const float*)d_in[17];
  const float* ln2_g    = (const float*)d_in[18];
  const float* ln2_b    = (const float*)d_in[19];
  const float* fc1_w    = (const float*)d_in[20];
  const float* fc1_b    = (const float*)d_in[21];
  const float* ln3_g    = (const float*)d_in[22];
  const float* ln3_b    = (const float*)d_in[23];
  const float* fc2_w    = (const float*)d_in[24];
  const float* fc2_b    = (const float*)d_in[25];
  const unsigned char* pad = (const unsigned char*)d_in[26];
  float* out = (float*)d_out;

  // ---- workspace carve-out ----
  char* ws = (char*)d_ws;
  size_t off = 0;
  auto alloc = [&](size_t bytes) { char* p = ws + off; off += align_up(bytes); return p; };
  half_t* wp_off1 = (half_t*)alloc((size_t)Ec * Ec * 2);
  half_t* wp_q    = (half_t*)alloc((size_t)Ec * Ec * 2);
  half_t* wp_k    = (half_t*)alloc((size_t)Ec * Ec * 2);
  half_t* wp_v    = (half_t*)alloc((size_t)Ec * Ec * 2);
  half_t* wp_o    = (half_t*)alloc((size_t)Ec * Ec * 2);
  half_t* wp_fc1  = (half_t*)alloc((size_t)Ec * 2 * Ec * 2);
  half_t* wp_fc2  = (half_t*)alloc((size_t)2 * Ec * Ec * 2);
  half_t* xn16    = (half_t*)alloc((size_t)Mtok * Ec * 2);
  half_t* h16     = (half_t*)alloc((size_t)Mtok * Ec * 2);
  half_t* q16     = (half_t*)alloc((size_t)Mtok * Ec * 2);
  float*  coords  = (float*)alloc((size_t)Mtok * Sc * 2 * 4);
  float*  dist    = (float*)alloc((size_t)Mtok * Sc * 4);
  half_t* samp16  = (half_t*)alloc((size_t)Msmp * Ec * 2);
  half_t* k16     = (half_t*)alloc((size_t)Msmp * Ec * 2);
  half_t* v16     = (half_t*)alloc((size_t)Msmp * Ec * 2);
  half_t* attn16  = (half_t*)alloc((size_t)Mtok * Ec * 2);
  float*  y32     = (float*)alloc((size_t)Mtok * Ec * 4);
  half_t* ln2_16  = (half_t*)alloc((size_t)Mtok * Ec * 2);
  float*  t32     = (float*)alloc((size_t)Mtok * 2 * Ec * 4);
  half_t* ln3_16  = (half_t*)alloc((size_t)Mtok * 2 * Ec * 2);

  // ---- 1. pack weights into WMMA fragment layout ----
  auto pack = [&](const float* w, half_t* wp, int K, int N) {
    int total = K * N;
    pack_weight_kernel<<<(total + 255) / 256, 256, 0, stream>>>(w, wp, K, N);
  };
  pack(off_w1, wp_off1, Ec, Ec);
  pack(q_w,    wp_q,    Ec, Ec);
  pack(k_w,    wp_k,    Ec, Ec);
  pack(v_w,    wp_v,    Ec, Ec);
  pack(o_w,    wp_o,    Ec, Ec);
  pack(fc1_w,  wp_fc1,  Ec, 2 * Ec);
  pack(fc2_w,  wp_fc2,  2 * Ec, Ec);

  auto gemm = [&](const half_t* A, const half_t* Wp, const float* bias,
                  float* o32, half_t* o16, const float* res,
                  const unsigned char* pd, int M, int N, int K, int mode) {
    int wtiles = (M >> 4) * (N >> 6);   // wave tile = 16x64
    gemm_wmma_kernel<<<(wtiles + 7) / 8, 256, 0, stream>>>(A, Wp, bias, o32, o16,
                                                           res, pd, M, N, K, mode);
  };

  // ---- 2. LN1 ----
  layernorm_kernel<<<Mtok, Ec, 0, stream>>>(x, ln1_g, ln1_b, xn16, Ec);
  // ---- 3. offset MLP layer 1 (gelu) ----
  gemm(xn16, wp_off1, off_b1, nullptr, h16, nullptr, nullptr, Mtok, Ec, Ec, 1);
  // ---- 4. offsets -> sample coords + dist bias ----
  offsets_kernel<<<(Mtok * Sc + 255) / 256, 256, 0, stream>>>(
      h16, off_w2, off_b2, base, coords, dist);
  // ---- 5. bilinear grid sample ----
  grid_sample_kernel<<<Mtok, Ec, 0, stream>>>(grid, coords, samp16);
  // ---- 6. q/k/v projections ----
  gemm(xn16,   wp_q, q_b, nullptr, q16, nullptr, nullptr, Mtok, Ec, Ec, 0);
  gemm(samp16, wp_k, k_b, nullptr, k16, nullptr, nullptr, Msmp, Ec, Ec, 0);
  gemm(samp16, wp_v, v_b, nullptr, v16, nullptr, nullptr, Msmp, Ec, Ec, 0);
  // ---- 7. deformable attention (softmax over S=8) ----
  attn_kernel<<<Mtok, Ec, 0, stream>>>(q16, k16, v16, dist, dist_pen, pad, attn16);
  // ---- 8. output projection + residual (pad-masked) ----
  gemm(attn16, wp_o, o_b, y32, nullptr, x, pad, Mtok, Ec, Ec, 2);
  // ---- 9. LN2 ----
  layernorm_kernel<<<Mtok, Ec, 0, stream>>>(y32, ln2_g, ln2_b, ln2_16, Ec);
  // ---- 10. fc1 + gelu ----
  gemm(ln2_16, wp_fc1, fc1_b, t32, nullptr, nullptr, nullptr, Mtok, 2 * Ec, Ec, 1);
  // ---- 11. LN3 (over 512) ----
  layernorm_kernel<<<Mtok, 2 * Ec, 0, stream>>>(t32, ln3_g, ln3_b, ln3_16, 2 * Ec);
  // ---- 12. fc2 + residual + pad zero -> output ----
  gemm(ln3_16, wp_fc2, fc2_b, out, nullptr, y32, pad, Mtok, Ec, 2 * Ec, 3);
}
